// MHA_9337258902078
// MI455X (gfx1250) — matrix operand order repaired and checked by reading, not compiled, not run on hardware
//
#include <hip/hip_runtime.h>
#include <hip/hip_bf16.h>

typedef __attribute__((ext_vector_type(16))) _Float16 v16h;
typedef __attribute__((ext_vector_type(8)))  _Float16 v8h;
typedef __attribute__((ext_vector_type(8)))  float    v8f;
typedef unsigned int u32x4 __attribute__((ext_vector_type(4)));
typedef int          i32x4 __attribute__((ext_vector_type(4)));
typedef int          i32x8 __attribute__((ext_vector_type(8)));

#define B_   2
#define S_   2048
#define D_   1024
#define H_   16
#define HD_  64
#define NTOK (B_ * S_)   // 4096 token rows

// ---------------------------------------------------------------------------
// WMMA helper: D = A(16x32 f16) * B(32x16 f16) + C(16x16 f32)
// ---------------------------------------------------------------------------
static __device__ __forceinline__ v8f wmma_f16(v16h a, v16h b, v8f c) {
  return __builtin_amdgcn_wmma_f32_16x16x32_f16(false, a, false, b,
                                                (short)0, c, false, false);
}

// A-operand (16x32, MxK): lane L holds row M=L%16; lg = L/16.
// VGPRs 0-3 hold K = 8*lg + 0..7, VGPRs 4-7 hold K = 16 + 8*lg + 0..7.
static __device__ __forceinline__ v16h load_a16(const _Float16* p, int lg) {
  union { v16h v; struct { v8h lo, hi; } s; } u;
  u.s.lo = *(const v8h*)(p + 8 * lg);
  u.s.hi = *(const v8h*)(p + 16 + 8 * lg);
  return u.v;
}

// B-operand (32x16, KxN): lane L holds column N=L%16;
// VGPRs 0-3 hold K = 8*lg + 0..7, VGPRs 4-7 hold K = 16 + 8*lg + 0..7.
static __device__ __forceinline__ v16h load_b16(const _Float16* p, int lg) {
  union { v16h v; struct { v8h lo, hi; } s; } u;
  u.s.lo = *(const v8h*)(p + 8 * lg);
  u.s.hi = *(const v8h*)(p + 16 + 8 * lg);
  return u.v;
}

// ---------------------------------------------------------------------------
// Tensor Data Mover: 2-D tile load (global -> LDS), f16 elements.
//   D# group0: count=1 | lds byte addr | 57-bit global addr | type=2
//   D# group1: data_size=2B; tensor dims == tile dims (no clipping);
//              tile_dim0 = row elems (contiguous), tile_dim1 = rows;
//              tensor_dim0_stride = row stride (elems). Groups 2/3 = 0 (2-D).
// ---------------------------------------------------------------------------
static __device__ __forceinline__ void tdm_load_2d(unsigned lds_byte_off,
                                                   const _Float16* gptr,
                                                   unsigned tile_w_elems,
                                                   unsigned tile_h_rows,
                                                   unsigned row_stride_elems) {
  const unsigned long long ga = (unsigned long long)(const void*)gptr;
  u32x4 g0;
  g0.x = 1u;                                                  // count=1
  g0.y = lds_byte_off;                                        // LDS dest
  g0.z = (unsigned)ga;                                        // gaddr[31:0]
  g0.w = (unsigned)((ga >> 32) & 0x01FFFFFFu) | (2u << 30);   // [56:32]|type=2
  i32x8 g1;
  g1[0] = (int)(1u << 16);                               // data_size=1 -> 2B
  g1[1] = (int)((tile_w_elems & 0xFFFFu) << 16);         // tensor_dim0 lo
  g1[2] = (int)((tile_h_rows & 0xFFFFu) << 16);          // dim0 hi=0 | dim1 lo
  g1[3] = (int)((tile_w_elems & 0xFFFFu) << 16);         // dim1 hi=0 | tile_dim0
  g1[4] = (int)(tile_h_rows & 0xFFFFu);                  // tile_dim1, tile_dim2=0
  g1[5] = (int)row_stride_elems;                         // dim0_stride[31:0]
  g1[6] = 0;                                             // stride hi / dim1_stride
  g1[7] = 0;
  const i32x4 zz = {0, 0, 0, 0};
#if defined(__clang_major__) && __clang_major__ >= 23
  const i32x8 z8 = {0, 0, 0, 0, 0, 0, 0, 0};
  __builtin_amdgcn_tensor_load_to_lds(g0, g1, zz, zz, z8, 0);
#else
  __builtin_amdgcn_tensor_load_to_lds(g0, g1, zz, zz, 0);
#endif
}

static __device__ __forceinline__ unsigned lds_off(const void* p) {
  return (unsigned)(unsigned long long)p;   // low 32 bits = LDS byte offset
}

// ---------------------------------------------------------------------------
// Kernel 0: f32 -> f16 convert
// ---------------------------------------------------------------------------
__global__ void cvt_f32_f16(const float* __restrict__ src,
                            _Float16* __restrict__ dst, int n) {
  int i = blockIdx.x * blockDim.x + threadIdx.x;
  if (i < n) dst[i] = (_Float16)src[i];
}

// ---------------------------------------------------------------------------
// Kernel 1: Y = X * W^T + bias. Block = 4 waves sharing one 64-col W tile
// (TDM double-buffered in LDS); each wave owns a 64-row strip.
//   vmode==0: Y[row, col] f16 (scaled); vmode==1: V transposed Vt[B][H][HD][S]
// ---------------------------------------------------------------------------
__global__ __launch_bounds__(128, 1)
void qkv_gemm(const _Float16* __restrict__ X, const _Float16* __restrict__ W,
              const float* __restrict__ bias, _Float16* __restrict__ Y,
              float scale, int vmode) {
  const int lane = threadIdx.x & 31;
  const int wslot = threadIdx.x >> 5;
  const int lg = lane >> 4, lr = lane & 15;
  const int tN = (blockIdx.x & 15) * 64;                 // 16 col tiles
  const int tM = (blockIdx.x >> 4) * 256 + wslot * 64;   // 16 blocks * 256 rows

  __shared__ __align__(32) _Float16 wtile[2][64 * 32];   // [row n][k] per buf

  if (wslot == 0)
    tdm_load_2d(lds_off(&wtile[0][0]), W + (size_t)tN * D_, 32, 64, D_);

  v8f acc[4][4];
  const v8f vzero = {};
#pragma unroll
  for (int mt = 0; mt < 4; ++mt)
#pragma unroll
    for (int nt = 0; nt < 4; ++nt) acc[mt][nt] = vzero;

  int cur = 0;
  for (int k = 0; k < D_; k += 32) {
    if (wslot == 0) {
      if (k + 32 < D_) {
        tdm_load_2d(lds_off(&wtile[cur ^ 1][0]),
                    W + (size_t)tN * D_ + k + 32, 32, 64, D_);
        __builtin_amdgcn_s_wait_tensorcnt(1);
      } else {
        __builtin_amdgcn_s_wait_tensorcnt(0);
      }
    }
    __syncthreads();

    v16h a[4], b[4];
#pragma unroll
    for (int mt = 0; mt < 4; ++mt)
      a[mt] = load_a16(X + (size_t)(tM + mt * 16 + lr) * D_ + k, lg);
#pragma unroll
    for (int nt = 0; nt < 4; ++nt)
      b[nt] = load_b16(&wtile[cur][(nt * 16 + lr) * 32], lg);
#pragma unroll
    for (int mt = 0; mt < 4; ++mt)
#pragma unroll
      for (int nt = 0; nt < 4; ++nt)
        acc[mt][nt] = wmma_f16(a[mt], b[nt], acc[mt][nt]);

    __syncthreads();
    cur ^= 1;
  }

#pragma unroll
  for (int nt = 0; nt < 4; ++nt) {
    const int col = tN + nt * 16 + lr;
    const float bn = bias[col];
#pragma unroll
    for (int mt = 0; mt < 4; ++mt) {
#pragma unroll
      for (int i = 0; i < 8; ++i) {
        const int row = tM + mt * 16 + lg * 8 + i;   // C layout: M = i + 8*lg
        const float v = (acc[mt][nt][i] + bn) * scale;
        if (vmode == 0) {
          Y[(size_t)row * D_ + col] = (_Float16)v;
        } else {  // transposed V: Vt[b][h][hd][s]
          const int bb = row / S_, t = row % S_;
          const int h = col / HD_, hd = col % HD_;
          Y[((size_t)(bb * H_ + h) * HD_ + hd) * S_ + t] = (_Float16)v;
        }
      }
    }
  }
}

// ---------------------------------------------------------------------------
// Kernel 2: flash attention. Block = 4 waves sharing (b, h): K/V tiles are
// TDM double-buffered in LDS once per block; each wave owns one 64-query tile.
//   Qh: [NTOK, D] f16 pre-scaled by 1/8.  Kh: [NTOK, D] f16.
//   Vt: [B][H][HD][S] f16.  Oh: [NTOK, D] f16.
// ---------------------------------------------------------------------------
#define PSTR 40   // padded LDS row stride (halfs) for the P round-trip

__global__ __launch_bounds__(128, 1)
void flash_attn(const _Float16* __restrict__ Qh, const _Float16* __restrict__ Kh,
                const _Float16* __restrict__ Vt, _Float16* __restrict__ Oh) {
  const int lane = threadIdx.x & 31;
  const int wslot = threadIdx.x >> 5;
  const int lg = lane >> 4, lr = lane & 15;
  const int qg = blockIdx.x & 7;            // 8 groups of 4 query tiles
  const int h  = (blockIdx.x >> 3) & 15;
  const int b  = blockIdx.x >> 7;
  const int qt = qg * 4 + wslot;            // this wave's 64-row query tile

  __shared__ __align__(32) _Float16 ktile[2][32 * 64];  // [key][hd]
  __shared__ __align__(32) _Float16 vtile[2][64 * 32];  // [hd][key]
  __shared__ __align__(32) _Float16 pbuf[4][64 * PSTR];
  _Float16* P = pbuf[wslot];

  const _Float16* Qbase = Qh + (size_t)(b * S_ + qt * 64) * D_ + h * HD_;
  const _Float16* Kbase = Kh + (size_t)(b * S_) * D_ + h * HD_;
  const _Float16* Vbase = Vt + (size_t)(b * H_ + h) * HD_ * S_;

  if (wslot == 0) {
    tdm_load_2d(lds_off(&ktile[0][0]), Kbase, 64, 32, D_);
    tdm_load_2d(lds_off(&vtile[0][0]), Vbase, 32, 64, S_);
  }

  // Preload Q as A-operands: 4 M-tiles x 2 k-steps (K = HD = 64)
  v16h aq[4][2];
#pragma unroll
  for (int mt = 0; mt < 4; ++mt)
#pragma unroll
    for (int ks = 0; ks < 2; ++ks)
      aq[mt][ks] = load_a16(Qbase + (size_t)(mt * 16 + lr) * D_ + ks * 32, lg);

  v8f o[4][4];
  v8f mrow[4], lrow[4];
  const v8f vzero = {};
#pragma unroll
  for (int mt = 0; mt < 4; ++mt) {
#pragma unroll
    for (int nt = 0; nt < 4; ++nt) o[mt][nt] = vzero;
#pragma unroll
    for (int i = 0; i < 8; ++i) { mrow[mt][i] = -3.0e38f; lrow[mt][i] = 0.0f; }
  }

  int cur = 0;
  for (int kt = 0; kt < S_; kt += 32) {
    if (wslot == 0) {
      if (kt + 32 < S_) {
        tdm_load_2d(lds_off(&ktile[cur ^ 1][0]),
                    Kbase + (size_t)(kt + 32) * D_, 64, 32, D_);
        tdm_load_2d(lds_off(&vtile[cur ^ 1][0]),
                    Vbase + (kt + 32), 32, 64, S_);
        __builtin_amdgcn_s_wait_tensorcnt(2);
      } else {
        __builtin_amdgcn_s_wait_tensorcnt(0);
      }
    }
    __syncthreads();

    // ---- scores S = Q * K^T (Q pre-scaled) : 64 x 32 tile ----
    v16h bk[2][2];
#pragma unroll
    for (int nt = 0; nt < 2; ++nt)
#pragma unroll
      for (int ks = 0; ks < 2; ++ks)
        bk[nt][ks] = load_b16(&ktile[cur][(nt * 16 + lr) * 64 + ks * 32], lg);

    v8f s[4][2];
#pragma unroll
    for (int mt = 0; mt < 4; ++mt) {
      s[mt][0] = vzero; s[mt][1] = vzero;
#pragma unroll
      for (int nt = 0; nt < 2; ++nt)
#pragma unroll
        for (int ks = 0; ks < 2; ++ks)
          s[mt][nt] = wmma_f16(aq[mt][ks], bk[nt][ks], s[mt][nt]);
    }

    // ---- online softmax: rows live across 16 lanes in C layout ----
#pragma unroll
    for (int mt = 0; mt < 4; ++mt) {
#pragma unroll
      for (int i = 0; i < 8; ++i) {
        float v = fmaxf(s[mt][0][i], s[mt][1][i]);
        v = fmaxf(v, __shfl_xor(v, 1));
        v = fmaxf(v, __shfl_xor(v, 2));
        v = fmaxf(v, __shfl_xor(v, 4));
        v = fmaxf(v, __shfl_xor(v, 8));
        const float mold = mrow[mt][i];
        const float mnew = fmaxf(mold, v);
        const float alpha = __expf(mold - mnew);
        mrow[mt][i] = mnew;
        const float p0 = __expf(s[mt][0][i] - mnew);
        const float p1 = __expf(s[mt][1][i] - mnew);
        s[mt][0][i] = p0; s[mt][1][i] = p1;
        float rs = p0 + p1;
        rs += __shfl_xor(rs, 1);
        rs += __shfl_xor(rs, 2);
        rs += __shfl_xor(rs, 4);
        rs += __shfl_xor(rs, 8);
        lrow[mt][i] = lrow[mt][i] * alpha + rs;
#pragma unroll
        for (int nt2 = 0; nt2 < 4; ++nt2) o[mt][nt2][i] *= alpha;
      }
      // spill P tile (f32 C-layout -> f16) to this wave's LDS slice
#pragma unroll
      for (int nt = 0; nt < 2; ++nt)
#pragma unroll
        for (int i = 0; i < 8; ++i)
          P[(mt * 16 + lg * 8 + i) * PSTR + nt * 16 + lr] =
              (_Float16)s[mt][nt][i];
    }

    // same-wave LDS ops are in-order (DScnt): read back as A-operand
    v16h ap[4];
#pragma unroll
    for (int mt = 0; mt < 4; ++mt)
      ap[mt] = load_a16(P + (mt * 16 + lr) * PSTR, lg);

    // ---- O += P * V : V columns (over keys) contiguous in vtile rows ----
#pragma unroll
    for (int nt2 = 0; nt2 < 4; ++nt2) {
      const v16h bv = load_b16(&vtile[cur][(nt2 * 16 + lr) * 32], lg);
#pragma unroll
      for (int mt = 0; mt < 4; ++mt)
        o[mt][nt2] = wmma_f16(ap[mt], bv, o[mt][nt2]);
    }

    __syncthreads();
    cur ^= 1;
  }

  // ---- finalize: O /= l, store f16 [token, D] ----
  _Float16* Obase = Oh + (size_t)(b * S_ + qt * 64) * D_ + h * HD_;
#pragma unroll
  for (int mt = 0; mt < 4; ++mt)
#pragma unroll
    for (int nt2 = 0; nt2 < 4; ++nt2)
#pragma unroll
      for (int i = 0; i < 8; ++i) {
        const float v = o[mt][nt2][i] / lrow[mt][i];
        Obase[(size_t)(mt * 16 + lg * 8 + i) * D_ + nt2 * 16 + lr] =
            (_Float16)v;
      }
}

// ---------------------------------------------------------------------------
// Kernel 3: out = attn * Wo^T + bo  (f32 output), same TDM staging as qkv_gemm
// ---------------------------------------------------------------------------
__global__ __launch_bounds__(128, 1)
void out_gemm(const _Float16* __restrict__ A, const _Float16* __restrict__ W,
              const float* __restrict__ bias, float* __restrict__ Y) {
  const int lane = threadIdx.x & 31;
  const int wslot = threadIdx.x >> 5;
  const int lg = lane >> 4, lr = lane & 15;
  const int tN = (blockIdx.x & 15) * 64;
  const int tM = (blockIdx.x >> 4) * 256 + wslot * 64;

  __shared__ __align__(32) _Float16 wtile[2][64 * 32];

  if (wslot == 0)
    tdm_load_2d(lds_off(&wtile[0][0]), W + (size_t)tN * D_, 32, 64, D_);

  v8f acc[4][4];
  const v8f vzero = {};
#pragma unroll
  for (int mt = 0; mt < 4; ++mt)
#pragma unroll
    for (int nt = 0; nt < 4; ++nt) acc[mt][nt] = vzero;

  int cur = 0;
  for (int k = 0; k < D_; k += 32) {
    if (wslot == 0) {
      if (k + 32 < D_) {
        tdm_load_2d(lds_off(&wtile[cur ^ 1][0]),
                    W + (size_t)tN * D_ + k + 32, 32, 64, D_);
        __builtin_amdgcn_s_wait_tensorcnt(1);
      } else {
        __builtin_amdgcn_s_wait_tensorcnt(0);
      }
    }
    __syncthreads();

    v16h a[4], b[4];
#pragma unroll
    for (int mt = 0; mt < 4; ++mt)
      a[mt] = load_a16(A + (size_t)(tM + mt * 16 + lr) * D_ + k, lg);
#pragma unroll
    for (int nt = 0; nt < 4; ++nt)
      b[nt] = load_b16(&wtile[cur][(nt * 16 + lr) * 32], lg);
#pragma unroll
    for (int mt = 0; mt < 4; ++mt)
#pragma unroll
      for (int nt = 0; nt < 4; ++nt)
        acc[mt][nt] = wmma_f16(a[mt], b[nt], acc[mt][nt]);

    __syncthreads();
    cur ^= 1;
  }

#pragma unroll
  for (int nt = 0; nt < 4; ++nt) {
    const int col = tN + nt * 16 + lr;
    const float bn = bias[col];
#pragma unroll
    for (int mt = 0; mt < 4; ++mt)
#pragma unroll
      for (int i = 0; i < 8; ++i) {
        const int row = tM + mt * 16 + lg * 8 + i;
        Y[(size_t)row * D_ + col] = acc[mt][nt][i] + bn;
      }
  }
}

// ---------------------------------------------------------------------------
// Host launcher
// ---------------------------------------------------------------------------
extern "C" void kernel_launch(void* const* d_in, const int* in_sizes, int n_in,
                              void* d_out, int out_size, void* d_ws,
                              size_t ws_size, hipStream_t stream) {
  const float* x  = (const float*)d_in[0];
  const float* Wq = (const float*)d_in[1];
  const float* bq = (const float*)d_in[2];
  const float* Wk = (const float*)d_in[3];
  const float* bk = (const float*)d_in[4];
  const float* Wv = (const float*)d_in[5];
  const float* bv = (const float*)d_in[6];
  const float* Wo = (const float*)d_in[7];
  const float* bo = (const float*)d_in[8];
  float* out = (float*)d_out;

  char* ws = (char*)d_ws;
  size_t off = 0;
  auto alloc = [&](size_t bytes) -> char* {
    char* p = ws + off;
    off += (bytes + 255) & ~(size_t)255;
    return p;
  };
  _Float16* xh  = (_Float16*)alloc((size_t)NTOK * D_ * 2);
  _Float16* wqh = (_Float16*)alloc((size_t)D_ * D_ * 2);
  _Float16* wkh = (_Float16*)alloc((size_t)D_ * D_ * 2);
  _Float16* wvh = (_Float16*)alloc((size_t)D_ * D_ * 2);
  _Float16* woh = (_Float16*)alloc((size_t)D_ * D_ * 2);
  _Float16* Qh  = (_Float16*)alloc((size_t)NTOK * D_ * 2);
  _Float16* Kh  = (_Float16*)alloc((size_t)NTOK * D_ * 2);
  _Float16* Vth = (_Float16*)alloc((size_t)NTOK * D_ * 2);  // [B,H,HD,S]
  _Float16* Ah  = (_Float16*)alloc((size_t)NTOK * D_ * 2);  // attn output

  const int nx = NTOK * D_;     // 4194304
  const int nw = D_ * D_;       // 1048576
  cvt_f32_f16<<<nx / 256, 256, 0, stream>>>(x, xh, nx);
  cvt_f32_f16<<<nw / 256, 256, 0, stream>>>(Wq, wqh, nw);
  cvt_f32_f16<<<nw / 256, 256, 0, stream>>>(Wk, wkh, nw);
  cvt_f32_f16<<<nw / 256, 256, 0, stream>>>(Wv, wvh, nw);
  cvt_f32_f16<<<nw / 256, 256, 0, stream>>>(Wo, woh, nw);

  // GEMMs: 256 blocks x 128 threads; block = 256 rows x 64 cols
  const dim3 gemmGrid(256), gemmBlock(128);
  qkv_gemm<<<gemmGrid, gemmBlock, 0, stream>>>(xh, wqh, bq, Qh, 0.125f, 0);
  qkv_gemm<<<gemmGrid, gemmBlock, 0, stream>>>(xh, wkh, bk, Kh, 1.0f, 0);
  qkv_gemm<<<gemmGrid, gemmBlock, 0, stream>>>(xh, wvh, bv, Vth, 1.0f, 1);

  // Attention: 256 blocks; block = (b, h, 4 query tiles)
  flash_attn<<<gemmGrid, gemmBlock, 0, stream>>>(Qh, Kh, Vth, Ah);

  out_gemm<<<gemmGrid, gemmBlock, 0, stream>>>(Ah, woh, bo, out);
}
